// VectorQuantizer_28604482191573
// MI455X (gfx1250) — matrix operand order repaired and checked by reading, not compiled, not run on hardware
//
#include <hip/hip_runtime.h>
#include <hip/hip_bf16.h>

typedef __bf16 bf16;
typedef __attribute__((ext_vector_type(16))) bf16  v16bf;
typedef __attribute__((ext_vector_type(8)))  bf16  v8bf;
typedef __attribute__((ext_vector_type(4)))  bf16  v4bf;
typedef __attribute__((ext_vector_type(8)))  float v8f;
typedef int v4i __attribute__((vector_size(16)));   // matches builtin param type

#define K_DIM 512
#define N_ROWS 32768
#define N_E 8192
#define ROWS_PER_BLOCK 64          // 4 waves x 16 rows, each wave owns a row tile
#define NT 16                      // codebook columns per LDS tile
#define NTILES (N_E / NT)          // 512 tiles
#define B_STRIDE 520               // 512+8 pad: 260 dwords ≡ 4 (mod 64) -> 2-per-bank b128

#define GLOBAL_AS __attribute__((address_space(1)))
#define LDS_AS    __attribute__((address_space(3)))

// 16-byte global -> LDS copy: async on CDNA5 toolchains that expose the builtin
__device__ __forceinline__ void copy16_g2l(void* lds, const void* g)
{
#if __has_builtin(__builtin_amdgcn_global_load_async_to_lds_b128)
    __builtin_amdgcn_global_load_async_to_lds_b128(
        (GLOBAL_AS v4i*)g, (LDS_AS v4i*)lds, 0, 0);
#else
    *reinterpret_cast<uint4*>(lds) = *reinterpret_cast<const uint4*>(g);
#endif
}

__device__ __forceinline__ void wait_async_copies()
{
#if __has_builtin(__builtin_amdgcn_global_load_async_to_lds_b128)
#if __has_builtin(__builtin_amdgcn_s_wait_asynccnt)
    __builtin_amdgcn_s_wait_asynccnt(0);
#else
    asm volatile("s_wait_asynccnt 0" ::: "memory");
#endif
#endif
}

// ---------------------------------------------------------------------------
// Row L2-normalize: emits bf16 copy (WMMA operand), optional f32 copy, 1/norm
// ---------------------------------------------------------------------------
__global__ __launch_bounds__(128) void vq_norm_rows(
    const float* __restrict__ X, bf16* __restrict__ Xb,
    float* __restrict__ Xf, float* __restrict__ inv_out)
{
    const int row = blockIdx.x;
    const int tid = threadIdx.x;                       // 128 threads * 4 = 512 elems
    const float4 v = reinterpret_cast<const float4*>(X + (size_t)row * K_DIM)[tid];
    float s = v.x * v.x + v.y * v.y + v.z * v.z + v.w * v.w;
#pragma unroll
    for (int m = 16; m >= 1; m >>= 1) s += __shfl_xor(s, m, 32);
    __shared__ float sred[4];
    if ((tid & 31) == 0) sred[tid >> 5] = s;
    __syncthreads();
    const float inv = rsqrtf(sred[0] + sred[1] + sred[2] + sred[3] + 1e-12f);
    float4 y;
    y.x = v.x * inv; y.y = v.y * inv; y.z = v.z * inv; y.w = v.w * inv;
    v4bf b;
    b[0] = (bf16)y.x; b[1] = (bf16)y.y; b[2] = (bf16)y.z; b[3] = (bf16)y.w;
    reinterpret_cast<v4bf*>(Xb + (size_t)row * K_DIM)[tid] = b;
    if (Xf) reinterpret_cast<float4*>(Xf + (size_t)row * K_DIM)[tid] = y;
    if (inv_out && tid == 0) inv_out[row] = inv;
}

// ---------------------------------------------------------------------------
// Fused bf16 WMMA GEMM (z_norm . emb_norm^T) + row argmax.
// Block = 128 threads / 4 waves, 64 z-rows. Wave w holds its 16-row A panel in
// 128 VGPRs (16 x v16bf) and scans ALL 8192 codebook columns. B tiles (16 cols
// x 512 K) are double-buffered in LDS, staged once per block with async
// global->LDS copies and consumed by all 4 waves (4x L2-traffic reuse).
// ---------------------------------------------------------------------------
__global__ __launch_bounds__(128) void vq_gemm_argmax(
    const bf16* __restrict__ zb, const bf16* __restrict__ eb,
    int* __restrict__ out_idx)
{
    __shared__ bf16 Bs[2][NT * B_STRIDE];              // 2 x 16.25 KB

    const int tid  = threadIdx.x;
    const int wave = tid >> 5;
    const int lane = tid & 31;
    const int half = lane >> 4;                        // A: K-offset 16*half; C: rows +8*half
    const int l16  = lane & 15;
    const int row0 = blockIdx.x * ROWS_PER_BLOCK;

    // --- A panel: 16 k-step fragments, loaded straight into registers -------
    // A layout used here: M = lane&15, K contiguous 16 at 16*half (matches B)
    const bf16* aRow = zb + (size_t)(row0 + wave * 16 + l16) * K_DIM + (half ? 16 : 0);
    v16bf afrag[16];
#pragma unroll
    for (int ks = 0; ks < 16; ++ks) {
        const v8bf lo = *reinterpret_cast<const v8bf*>(aRow + ks * 32);
        const v8bf hi = *reinterpret_cast<const v8bf*>(aRow + ks * 32 + 8);
        afrag[ks] = __builtin_shufflevector(lo, hi,
            0, 1, 2, 3, 4, 5, 6, 7, 8, 9, 10, 11, 12, 13, 14, 15);
    }

    float best[8];
    int   bidx[8];
#pragma unroll
    for (int r = 0; r < 8; ++r) { best[r] = -3.4e38f; bidx[r] = 0; }

    // --- B tile staging: 16 cols x 512 K = 1024 x 16B chunks, 8 per thread --
    auto stage = [&](int buf, int t) {
        const int nb = t * NT;
#pragma unroll
        for (int i = 0; i < 8; ++i) {
            const int c   = tid + 128 * i;             // 0..1023
            const int r   = c >> 6;                    // local col 0..15
            const int k8  = c & 63;                    // 8-elem chunk in K
            copy16_g2l(&Bs[buf][r * B_STRIDE + k8 * 8],
                       eb + (size_t)(nb + r) * K_DIM + k8 * 8);
        }
    };

    stage(0, 0);
    wait_async_copies();
    __syncthreads();

    for (int t = 0; t < NTILES; ++t) {
        const int cur = t & 1;
        if (t + 1 < NTILES) stage(cur ^ 1, t + 1);

        // B fragment base: N = lane&15 (local col), K contiguous 16 at 16*half
        const bf16* bsBase = &Bs[cur][l16 * B_STRIDE + (half ? 16 : 0)];

        // 4 independent K-split accumulator chains for XDL pipelining
        v8f acc[4];
#pragma unroll
        for (int a = 0; a < 4; ++a)
            acc[a] = (v8f){0.f, 0.f, 0.f, 0.f, 0.f, 0.f, 0.f, 0.f};
#pragma unroll
        for (int ks = 0; ks < 16; ++ks) {
            const v16bf bb = *reinterpret_cast<const v16bf*>(bsBase + ks * 32);
            acc[ks & 3] = __builtin_amdgcn_wmma_f32_16x16x32_bf16(
                false, afrag[ks], false, bb, (short)0, acc[ks & 3], false, false);
        }
        const v8f tot = acc[0] + acc[1] + acc[2] + acc[3];

        const int c0 = t * NT + l16;                   // global code index
#pragma unroll
        for (int r = 0; r < 8; ++r) {
            const float v = tot[r];
            if (v > best[r]) { best[r] = v; bidx[r] = c0; }
        }

        wait_async_copies();
        __syncthreads();
    }

    // Butterfly reduce within each 16-lane half (first-index tie-break).
    // Slot r of half h is row (wave*16 + 8h + r); each wave covered all cols.
#pragma unroll
    for (int m = 8; m >= 1; m >>= 1) {
#pragma unroll
        for (int r = 0; r < 8; ++r) {
            const float ov = __shfl_xor(best[r], m, 32);
            const int   oi = __shfl_xor(bidx[r], m, 32);
            if (ov > best[r] || (ov == best[r] && oi < bidx[r])) {
                best[r] = ov; bidx[r] = oi;
            }
        }
    }
    if (l16 == 0) {
#pragma unroll
        for (int r = 0; r < 8; ++r)
            out_idx[row0 + wave * 16 + half * 8 + r] = bidx[r];
    }
}

// ---------------------------------------------------------------------------
// z_q = emb_norm[idx]; per-row squared-error partial; indices as float
// ---------------------------------------------------------------------------
__global__ __launch_bounds__(128) void vq_gather_loss(
    const float* __restrict__ z, const float* __restrict__ inv_norm,
    const float* __restrict__ enf, const int* __restrict__ idx,
    float* __restrict__ out_zq, float* __restrict__ out_idx_f,
    float* __restrict__ partial)
{
    const int row = blockIdx.x;
    const int tid = threadIdx.x;
    const int e   = idx[row];
    const float4 q = reinterpret_cast<const float4*>(enf + (size_t)e * K_DIM)[tid];
    reinterpret_cast<float4*>(out_zq + (size_t)row * K_DIM)[tid] = q;
    const float inv = inv_norm[row];
    const float4 zv = reinterpret_cast<const float4*>(z + (size_t)row * K_DIM)[tid];
    const float dx = q.x - zv.x * inv;
    const float dy = q.y - zv.y * inv;
    const float dz = q.z - zv.z * inv;
    const float dw = q.w - zv.w * inv;
    float s = dx * dx + dy * dy + dz * dz + dw * dw;
#pragma unroll
    for (int m = 16; m >= 1; m >>= 1) s += __shfl_xor(s, m, 32);
    __shared__ float sred[4];
    if ((tid & 31) == 0) sred[tid >> 5] = s;
    __syncthreads();
    if (tid == 0) {
        partial[row]   = sred[0] + sred[1] + sred[2] + sred[3];
        out_idx_f[row] = (float)e;
    }
}

// Deterministic single-block scalar loss reduction
__global__ __launch_bounds__(256) void vq_reduce_loss(
    const float* __restrict__ partial, float* __restrict__ out_loss)
{
    __shared__ float sm[256];
    const int tid = threadIdx.x;
    float s = 0.f;
    for (int i = tid; i < N_ROWS; i += 256) s += partial[i];
    sm[tid] = s;
    __syncthreads();
    for (int m = 128; m >= 1; m >>= 1) {
        if (tid < m) sm[tid] += sm[tid + m];
        __syncthreads();
    }
    // loss = (BETA + 1) * mean = 1.25 * sum / (32768*512)
    if (tid == 0) out_loss[0] = sm[0] * (1.25f / 16777216.0f);
}

// ---------------------------------------------------------------------------
extern "C" void kernel_launch(void* const* d_in, const int* in_sizes, int n_in,
                              void* d_out, int out_size, void* d_ws, size_t ws_size,
                              hipStream_t stream)
{
    (void)in_sizes; (void)n_in; (void)out_size; (void)ws_size;
    const float* z   = (const float*)d_in[0];   // [8,4096,512] f32
    const float* emb = (const float*)d_in[1];   // [8192,512]  f32
    float* out = (float*)d_out;

    char* p = (char*)d_ws;
    bf16*  zbuf = (bf16*)p;  p += (size_t)N_ROWS * K_DIM * sizeof(bf16);   // 32 MB
    bf16*  ebuf = (bf16*)p;  p += (size_t)N_E    * K_DIM * sizeof(bf16);   // 8 MB
    float* enf  = (float*)p; p += (size_t)N_E    * K_DIM * sizeof(float);  // 16 MB
    float* invn = (float*)p; p += (size_t)N_ROWS * sizeof(float);
    int*   idx  = (int*)p;   p += (size_t)N_ROWS * sizeof(int);
    float* part = (float*)p; p += (size_t)N_ROWS * sizeof(float);

    float* out_zq   = out;                                   // 32768*512
    float* out_loss = out + (size_t)N_ROWS * K_DIM;          // 1
    float* out_idxf = out_loss + 1;                          // 32768

    vq_norm_rows<<<N_ROWS, 128, 0, stream>>>(z,   zbuf, nullptr, invn);
    vq_norm_rows<<<N_E,    128, 0, stream>>>(emb, ebuf, enf,     nullptr);
    vq_gemm_argmax<<<N_ROWS / ROWS_PER_BLOCK, 128, 0, stream>>>(zbuf, ebuf, idx);
    vq_gather_loss<<<N_ROWS, 128, 0, stream>>>(z, invn, enf, idx, out_zq, out_idxf, part);
    vq_reduce_loss<<<1, 256, 0, stream>>>(part, out_loss);
}
